// RNNModel_32160715112736
// MI455X (gfx1250) — matrix operand order repaired and checked by reading, not compiled, not run on hardware
//
#include <hip/hip_runtime.h>
#include <math.h>

#define BB 128
#define TT 512
#define HH 512
#define DIN1 128
#define DOUT 128
#define NWG 32          // H/16 column slices for the RNN layers

typedef __attribute__((ext_vector_type(16))) __bf16 v16bf;
typedef __attribute__((ext_vector_type(8)))  float  v8f;

union Frag {
    v16bf v;
    uint4 q[2];
};

__device__ __forceinline__ unsigned short f2bf(float f) {
    union { float f; unsigned u; } c; c.f = f;
    unsigned u = c.u;
    if ((u & 0x7fffffffu) > 0x7f800000u) return (unsigned short)0x7fc0; // NaN
    unsigned r = 0x7fffu + ((u >> 16) & 1u);                            // RNE
    return (unsigned short)((u + r) >> 16);
}

__device__ __forceinline__ float gelu_erf(float x) {
    return 0.5f * x * (1.0f + erff(x * 0.70710678118654752440f));
}

// A-matrix 16x32 bf16 fragment (row-major source, lane row = lane&15):
//   elems 0..7  = K [k0+8*kh,  k0+8*kh+8)
//   elems 8..15 = K [k0+16+8*kh, +8)
__device__ __forceinline__ v16bf load_a(const unsigned short* row_base, int k0, int kh) {
    Frag f;
    const unsigned short* p = row_base + k0 + 8 * kh;
    f.q[0] = *(const uint4*)(p);
    f.q[1] = *(const uint4*)(p + 16);
    return f.v;
}

// B-matrix 32x16 bf16 fragment: lane column base points at W[n][*] (K-major),
// elems 0..7 = K [k0+8*kh, +8), elems 8..15 = K [k0+16+8*kh, +8).
__device__ __forceinline__ v16bf load_b(const unsigned short* col_base, int k0, int kh) {
    Frag f;
    const unsigned short* p = col_base + k0 + 8 * kh;
    f.q[0] = *(const uint4*)(p);
    f.q[1] = *(const uint4*)(p + 16);
    return f.v;
}

__device__ __forceinline__ v8f wmma_bf16(v16bf a, v16bf b, v8f c) {
    return __builtin_amdgcn_wmma_f32_16x16x32_bf16(false, a, false, b, (short)0, c, false, false);
}

// Device-wide barrier: single counter + generation word (reset-before-bump).
__device__ void gsync(unsigned* bar, unsigned* gen, unsigned nwg) {
    __syncthreads();
    if (threadIdx.x == 0) {
        __threadfence();
        unsigned g   = atomicAdd(gen, 0u);
        unsigned old = atomicAdd(bar, 1u);
        if (old == nwg - 1u) {
            atomicExch(bar, 0u);     // reset BEFORE releasing spinners
            atomicAdd(gen, 1u);
        } else {
            while (atomicAdd(gen, 0u) == g) { __builtin_amdgcn_s_sleep(8); }
        }
        __threadfence();
    }
    __syncthreads();
}

// ---------------------------------------------------------------------------
// Prep: fp32 -> bf16 conversions + barrier init (ws is poisoned by harness).
// ---------------------------------------------------------------------------
__global__ __launch_bounds__(256) void prep_kernel(
    const float* __restrict__ X, const float* __restrict__ Wl,
    unsigned short* __restrict__ Xbf, unsigned short* __restrict__ Wlb,
    unsigned* bar, unsigned* gen)
{
    size_t gt = (size_t)blockIdx.x * blockDim.x + threadIdx.x;
    size_t gs = (size_t)gridDim.x * blockDim.x;
    const size_t nX = (size_t)BB * TT * DIN1;
    for (size_t i = gt; i < nX; i += gs) Xbf[i] = f2bf(X[i]);
    const size_t nW = (size_t)DOUT * HH;
    for (size_t i = gt; i < nW; i += gs) Wlb[i] = f2bf(Wl[i]);
    if (gt == 0) { *bar = 0u; *gen = 0u; }
}

// ---------------------------------------------------------------------------
// Persistent RNN layer: 32 WGs, WG n owns output columns [16n,16n+16).
// 8 waves/WG, wave w owns batch rows [16w,16w+16). One grid barrier / step.
// h_new = gelu([x_t,h] @ Wi^T + bi) ; o_t = gelu(h_new @ Wo^T + bo)
// ---------------------------------------------------------------------------
__global__ __launch_bounds__(256) void rnn_layer_kernel(
    const unsigned short* __restrict__ xin, int DIN,
    const float* __restrict__ Wi, const float* __restrict__ bi,
    const float* __restrict__ Wo, const float* __restrict__ bo,
    unsigned short* __restrict__ out,
    unsigned short* __restrict__ h0buf, unsigned short* __restrict__ h1buf,
    unsigned* bar, unsigned* gen)
{
    __shared__ __align__(16) unsigned short WiS[16 * 1024]; // 16 x (DIN+H) bf16
    __shared__ __align__(16) unsigned short WoS[16 * 512];  // 16 x H bf16

    const int Kw  = DIN + HH;
    const int n16 = blockIdx.x * 16;

    // Stage this WG's weight slice into LDS (fp32 -> bf16 once).
    for (int i = threadIdx.x; i < 16 * Kw; i += 256) {
        int r = i / Kw, c = i - r * Kw;
        WiS[r * Kw + c] = f2bf(Wi[(size_t)(n16 + r) * Kw + c]);
    }
    for (int i = threadIdx.x; i < 16 * HH; i += 256) {
        int r = i >> 9, c = i & (HH - 1);
        WoS[i] = f2bf(Wo[(size_t)(n16 + r) * HH + c]);
    }
    // h(0) = 0 (cooperative across the grid).
    {
        size_t gt = (size_t)blockIdx.x * 256 + threadIdx.x;
        size_t gs = (size_t)gridDim.x * 256;
        for (size_t i = gt; i < (size_t)BB * HH; i += gs) h0buf[i] = 0;
    }
    gsync(bar, gen, gridDim.x);

    const int wave = threadIdx.x >> 5;
    const int lane = threadIdx.x & 31;
    const int rn   = lane & 15;          // A row within tile == B column within slice
    const int kh   = lane >> 4;
    const int b    = wave * 16 + rn;     // batch row this lane reads for A
    const int ncol = n16 + rn;           // output column this lane produces
    const float bi_n = bi[ncol];
    const float bo_n = bo[ncol];
    const unsigned short* wi_col = &WiS[rn * Kw];
    const unsigned short* wo_col = &WoS[rn * HH];

    for (int t = 0; t < TT; ++t) {
        const unsigned short* hprev = (t & 1) ? h1buf : h0buf;
        unsigned short*       hnext = (t & 1) ? h0buf : h1buf;

        // ---- phase A: h_new slice = gelu([x_t, h_prev] @ Wi^T + bi) ----
        v8f acc = {};
        const unsigned short* xrow = xin + ((size_t)b * TT + t) * DIN;
        for (int k0 = 0; k0 < DIN; k0 += 32)
            acc = wmma_bf16(load_a(xrow, k0, kh), load_b(wi_col, k0, kh), acc);
        const unsigned short* hrow = hprev + (size_t)b * HH;
        for (int k0 = 0; k0 < HH; k0 += 32)
            acc = wmma_bf16(load_a(hrow, k0, kh), load_b(wi_col, DIN + k0, kh), acc);
#pragma unroll
        for (int r = 0; r < 8; ++r) {
            float hv = gelu_erf(acc[r] + bi_n);
            hnext[(size_t)(wave * 16 + r + 8 * kh) * HH + ncol] = f2bf(hv);
        }

        gsync(bar, gen, gridDim.x);   // publish full h_new to all WGs

        // ---- phase B: o_t slice = gelu(h_new @ Wo^T + bo) ----
        v8f acc2 = {};
        const unsigned short* hnrow = hnext + (size_t)b * HH;
        for (int k0 = 0; k0 < HH; k0 += 32)
            acc2 = wmma_bf16(load_a(hnrow, k0, kh), load_b(wo_col, k0, kh), acc2);
#pragma unroll
        for (int r = 0; r < 8; ++r) {
            float ov = gelu_erf(acc2[r] + bo_n);
            out[((size_t)(wave * 16 + r + 8 * kh) * TT + t) * HH + ncol] = f2bf(ov);
        }
        // No barrier needed here: next step writes the *other* h buffer, and
        // all reads of that buffer finished before the barrier above.
    }
}

// ---------------------------------------------------------------------------
// Final linear: [B*T,512] @ [512,128]^T + b. Block = 16 rows, wave = 1 N-tile.
// ---------------------------------------------------------------------------
__global__ __launch_bounds__(256) void final_gemm_kernel(
    const unsigned short* __restrict__ o2, const unsigned short* __restrict__ Wb,
    const float* __restrict__ blast, float* __restrict__ outp)
{
    const int mtile = blockIdx.x;
    const int wave  = threadIdx.x >> 5;   // 8 waves == 8 N-tiles (DOUT=128)
    const int lane  = threadIdx.x & 31;
    const int rn    = lane & 15;
    const int kh    = lane >> 4;

    const unsigned short* arow = o2 + (size_t)(mtile * 16 + rn) * HH;
    const unsigned short* brow = Wb + (size_t)(wave * 16 + rn) * HH;

    v8f acc = {};
    for (int k0 = 0; k0 < HH; k0 += 32)
        acc = wmma_bf16(load_a(arow, k0, kh), load_b(brow, k0, kh), acc);

    const float bias = blast[wave * 16 + rn];
#pragma unroll
    for (int r = 0; r < 8; ++r)
        outp[(size_t)(mtile * 16 + r + 8 * kh) * DOUT + wave * 16 + rn] = acc[r] + bias;
}

// ---------------------------------------------------------------------------
extern "C" void kernel_launch(void* const* d_in, const int* in_sizes, int n_in,
                              void* d_out, int out_size, void* d_ws, size_t ws_size,
                              hipStream_t stream) {
    const float* X   = (const float*)d_in[0];
    const float* Wi1 = (const float*)d_in[1];
    const float* bi1 = (const float*)d_in[2];
    const float* Wo1 = (const float*)d_in[3];
    const float* bo1 = (const float*)d_in[4];
    const float* Wi2 = (const float*)d_in[5];
    const float* bi2 = (const float*)d_in[6];
    const float* Wo2 = (const float*)d_in[7];
    const float* bo2 = (const float*)d_in[8];
    const float* Wl  = (const float*)d_in[9];
    const float* bl  = (const float*)d_in[10];
    float* outp = (float*)d_out;

    char* w = (char*)d_ws;
    size_t off = 0;
    unsigned short* Xbf = (unsigned short*)(w + off); off += (size_t)BB * TT * DIN1 * 2;
    unsigned short* o1  = (unsigned short*)(w + off); off += (size_t)BB * TT * HH * 2;
    unsigned short* o2  = (unsigned short*)(w + off); off += (size_t)BB * TT * HH * 2;
    unsigned short* h0  = (unsigned short*)(w + off); off += (size_t)BB * HH * 2;
    unsigned short* h1  = (unsigned short*)(w + off); off += (size_t)BB * HH * 2;
    unsigned short* Wlb = (unsigned short*)(w + off); off += (size_t)DOUT * HH * 2;
    unsigned* bar = (unsigned*)(w + off); off += 64;
    unsigned* gen = (unsigned*)(w + off); off += 64;

    prep_kernel<<<512, 256, 0, stream>>>(X, Wl, Xbf, Wlb, bar, gen);

    rnn_layer_kernel<<<NWG, 256, 0, stream>>>(Xbf, DIN1, Wi1, bi1, Wo1, bo1,
                                              o1, h0, h1, bar, gen);
    rnn_layer_kernel<<<NWG, 256, 0, stream>>>(o1, HH, Wi2, bi2, Wo2, bo2,
                                              o2, h0, h1, bar, gen);

    final_gemm_kernel<<<(BB * TT) / 16, 256, 0, stream>>>(o2, Wlb, bl, outp);
}
